// LatentGRUDecoderS_18253611008502
// MI455X (gfx1250) — matrix operand order repaired and checked by reading, not compiled, not run on hardware
//
#include <hip/hip_runtime.h>
#include <hip/hip_bf16.h>

// ---------------------------------------------------------------------------
// LatentGRUDecoder for gfx1250 (MI455X, wave32, WMMA)
//   B=128, T=512, H=512, NB=2, IN=3
//   Sequential GRU: 32 persistent WGs, each owns a 16-column slice of H,
//   global barrier per timestep.  bf16 WMMA 16x16x32, f32 accumulate.
//   MLP (H->H relu, H->1) fused into the step loop; deterministic reduction.
// ---------------------------------------------------------------------------

#define B_   128
#define T_   512
#define H_   512
#define G3H  1536
#define NWG  32          // H_/16 workgroups, each owns 16 columns of H

typedef __attribute__((ext_vector_type(16))) __bf16 v16bf;
typedef __attribute__((ext_vector_type(8)))  float  v8f;

union BF16x16 { v16bf v; uint4 q[2]; };

// A-matrix tile (16x32 bf16): lane L -> row = L%16 (caller picks row ptr),
// elements e = K = k0 + 16*(L/16) + e
__device__ __forceinline__ v16bf load_A(const __hip_bfloat16* rowbase, int k0, int kh) {
    const char* p = (const char*)(rowbase + k0 + 16 * kh);
    BF16x16 u;
    u.q[0] = *(const uint4*)(p);
    u.q[1] = *(const uint4*)(p + 16);   // +8 elements
    return u.v;
}

// B-matrix tile (32x16 bf16), weights stored transposed [N,K]:
// lane L -> col = L%16 (caller picks row ptr), elements e = K = k0 + 16*(L/16) + e
__device__ __forceinline__ v16bf load_B(const __hip_bfloat16* wrow, int k0, int kh) {
    const uint4* p = (const uint4*)(wrow + k0 + 16 * kh);
    BF16x16 u;
    u.q[0] = p[0];
    u.q[1] = p[1];
    return u.v;
}

__device__ __forceinline__ v8f wmma_bf16(v16bf a, v16bf b, v8f c) {
    return __builtin_amdgcn_wmma_f32_16x16x32_bf16(false, a, false, b, (short)0, c,
                                                   false, false);
}

// Fast sigmoid: single v_exp_f32 + v_rcp_f32 (avoid IEEE divide expansion).
__device__ __forceinline__ float fast_sigmoid(float x) {
    return __builtin_amdgcn_rcpf(1.0f + __expf(-x));
}

// Fast tanh: hardware TANH transcendental when available, else exp+rcp.
__device__ __forceinline__ float fast_tanh(float x) {
#if __has_builtin(__builtin_amdgcn_tanhf)
    return __builtin_amdgcn_tanhf(x);
#else
    // tanh(x) = 1 - 2/(exp(2x)+1); exp overflow -> inf -> rcp -> 0 -> +1 (ok),
    // exp underflow -> 0 -> 1 - 2 = -1 (ok).
    return 1.0f - 2.0f * __builtin_amdgcn_rcpf(__expf(2.0f * x) + 1.0f);
#endif
}

// ---------------------------------------------------------------------------
// Prep: zero barrier counters, transpose+convert weights to bf16, init h.
// ---------------------------------------------------------------------------
__global__ void prep_kernel(const float* __restrict__ encz,
                            const float* __restrict__ W_hh,
                            const float* __restrict__ W1,
                            float* __restrict__ hf32,
                            __hip_bfloat16* __restrict__ hbf0,
                            __hip_bfloat16* __restrict__ WtHH,
                            __hip_bfloat16* __restrict__ W1t,
                            int* __restrict__ ctr) {
    const size_t i0     = (size_t)blockIdx.x * blockDim.x + threadIdx.x;
    const size_t stride = (size_t)gridDim.x * blockDim.x;

    // WtHH[n][k] = W_hh[k][n]   (n in [0,3H), k in [0,H))
    for (size_t idx = i0; idx < (size_t)G3H * H_; idx += stride) {
        const size_t n = idx / H_, k = idx % H_;
        WtHH[idx] = __float2bfloat16(W_hh[k * G3H + n]);
    }
    // W1t[o][i] = W1[i][o]
    for (size_t idx = i0; idx < (size_t)H_ * H_; idx += stride) {
        const size_t o = idx / H_, ii = idx % H_;
        W1t[idx] = __float2bfloat16(W1[ii * H_ + o]);
    }
    // h init from encz_last
    for (size_t idx = i0; idx < (size_t)B_ * H_; idx += stride) {
        const float v = encz[idx];
        hf32[idx] = v;
        hbf0[idx] = __float2bfloat16(v);
    }
    // barrier counters (must be reset every call: graph replay)
    for (size_t idx = i0; idx < (size_t)T_; idx += stride) ctr[idx] = 0;
}

// ---------------------------------------------------------------------------
// Persistent GRU + fused MLP.  <<<NWG, 256>>>  (8 waves of 32)
//   wave w owns batch rows [16w, 16w+16); block g owns H-cols [16g, 16g+16)
// ---------------------------------------------------------------------------
__global__ void __launch_bounds__(256)
gru_kernel(const float* __restrict__ dtime,
           const unsigned char* __restrict__ s1,   // s_onehot [B,T,2] bool
           const unsigned char* __restrict__ oh,   // onehot   [B,T]   bool
           const float* __restrict__ W_ih,         // [3, 3H]
           const float* __restrict__ b_ih,         // [3H]
           const float* __restrict__ b_hh,         // [3H]
           const float* __restrict__ b1,           // [H]
           const float* __restrict__ W2,           // [H]
           const __hip_bfloat16* __restrict__ WtHH,  // [3H, H] (transposed)
           const __hip_bfloat16* __restrict__ W1t,   // [H, H]  (transposed)
           float* __restrict__ hf32,               // [B, H] master state
           __hip_bfloat16* __restrict__ hbf0,      // [B, H] ping
           __hip_bfloat16* __restrict__ hbf1,      // [B, H] pong
           float* __restrict__ partial,            // [T, NWG, B]
           int* __restrict__ ctr) {                // [T]
    const int tid  = threadIdx.x;
    const int lane = tid & 31;
    const int w    = tid >> 5;          // wave id 0..7  -> M tile
    const int g    = blockIdx.x;        // 0..31         -> N slice
    const int j0   = g << 4;
    const int col  = lane & 15;
    const int kh   = lane >> 4;         // 0/1 (K-half selector)
    const int jc   = j0 + col;          // this lane's output column in H

    // Per-lane column-constant parameters (L2-hot scalar loads, hoisted)
    float wih0[3], wihA[3], wihB[3], bih[3], bhh[3];
#pragma unroll
    for (int q = 0; q < 3; ++q) {
        const int gc = q * H_ + jc;
        wih0[q] = W_ih[0 * G3H + gc];
        wihA[q] = W_ih[1 * G3H + gc];
        wihB[q] = W_ih[2 * G3H + gc];
        bih[q]  = b_ih[gc];
        bhh[q]  = b_hh[gc];
    }
    const float b1c = b1[jc];
    const float w2c = W2[jc];

    const int arow = (w << 4) + col;    // A-matrix row for this lane
    const __hip_bfloat16* WrowR = WtHH + (size_t)(0 * H_ + jc) * H_;
    const __hip_bfloat16* WrowZ = WtHH + (size_t)(1 * H_ + jc) * H_;
    const __hip_bfloat16* WrowN = WtHH + (size_t)(2 * H_ + jc) * H_;
    const __hip_bfloat16* W1row = W1t  + (size_t)jc * H_;

    for (int t = 0; t < T_; ++t) {
        const __hip_bfloat16* hcur = (t & 1) ? hbf1 : hbf0;
        __hip_bfloat16*       hnxt = (t & 1) ? hbf0 : hbf1;

        // ---- hp = h @ W_hh : three 16x16 gate tiles, K=512 ----
        v8f cr = {}, cz = {}, cn = {};
        const __hip_bfloat16* arowp = hcur + (size_t)arow * H_;
#pragma unroll 2
        for (int k0 = 0; k0 < H_; k0 += 32) {
            const v16bf a  = load_A(arowp, k0, kh);
            const v16bf br = load_B(WrowR, k0, kh);
            const v16bf bz = load_B(WrowZ, k0, kh);
            const v16bf bn = load_B(WrowN, k0, kh);
            cr = wmma_bf16(a, br, cr);
            cz = wmma_bf16(a, bz, cz);
            cn = wmma_bf16(a, bn, cn);
        }

        // ---- gates + state update (f32) ----
#pragma unroll
        for (int e = 0; e < 8; ++e) {
            const int   b    = (w << 4) + (kh << 3) + e;   // C-tile row -> batch
            const size_t bt  = (size_t)b * T_ + t;
            const float dt   = dtime[bt];
            const int   band = s1[bt * 2 + 1] ? 1 : 0;
            const int   m    = oh[bt];
            const float hold = hf32[(size_t)b * H_ + jc];

            const float xr = dt * wih0[0] + (band ? wihB[0] : wihA[0]) + bih[0];
            const float xz = dt * wih0[1] + (band ? wihB[1] : wihA[1]) + bih[1];
            const float xn = dt * wih0[2] + (band ? wihB[2] : wihA[2]) + bih[2];

            const float r = fast_sigmoid(xr + (cr[e] + bhh[0]));
            const float z = fast_sigmoid(xz + (cz[e] + bhh[1]));
            const float n = fast_tanh(xn + r * (cn[e] + bhh[2]));
            const float hnew = m ? ((1.0f - z) * n + z * hold) : hold;

            hf32[(size_t)b * H_ + jc] = hnew;
            hnxt[(size_t)b * H_ + jc] = __float2bfloat16(hnew);
        }

        // ---- device-wide barrier (32 co-resident WGs) ----
        __threadfence();
        __syncthreads();
        if (tid == 0) {
            __hip_atomic_fetch_add(&ctr[t], 1, __ATOMIC_ACQ_REL,
                                   __HIP_MEMORY_SCOPE_AGENT);
            while (__hip_atomic_load(&ctr[t], __ATOMIC_ACQUIRE,
                                     __HIP_MEMORY_SCOPE_AGENT) < NWG) {
                __builtin_amdgcn_s_sleep(1);   // polite spin
            }
        }
        __syncthreads();
        __threadfence();

        // ---- fused MLP on decz[t] = hnxt : h1 = relu(h@W1+b1); p = h1.W2 ----
        v8f ca = {};
        const __hip_bfloat16* arow2 = hnxt + (size_t)arow * H_;
#pragma unroll 2
        for (int k0 = 0; k0 < H_; k0 += 32) {
            const v16bf a  = load_A(arow2, k0, kh);
            const v16bf bw = load_B(W1row, k0, kh);
            ca = wmma_bf16(a, bw, ca);
        }
#pragma unroll
        for (int e = 0; e < 8; ++e) {
            float h1 = ca[e] + b1c;
            h1 = h1 > 0.0f ? h1 : 0.0f;
            float pv = h1 * w2c;
            // reduce over the 16 lanes of this half-wave (distinct columns)
#pragma unroll
            for (int msk = 8; msk >= 1; msk >>= 1)
                pv += __shfl_xor(pv, msk, 32);
            if (col == 0) {
                const int b = (w << 4) + (kh << 3) + e;
                partial[((size_t)t * NWG + g) * B_ + b] = pv;
            }
        }
    }
}

// ---------------------------------------------------------------------------
// decx[b,t] = sum_g partial[t,g,b] + b2 ; also zero the output buffer.
// ---------------------------------------------------------------------------
__global__ void reduce_kernel(const float* __restrict__ partial,
                              const float* __restrict__ b2,
                              float* __restrict__ decx,
                              float* __restrict__ out) {
    const int i = blockIdx.x * blockDim.x + threadIdx.x;   // [0, B*T)
    if (i >= B_ * T_) return;
    const int b = i / T_, t = i % T_;
    float s = b2[0];
#pragma unroll
    for (int g = 0; g < NWG; ++g)
        s += partial[((size_t)t * NWG + g) * B_ + b];
    decx[(size_t)b * T_ + t] = s;
    out[i]            = 0.0f;
    out[B_ * T_ + i]  = 0.0f;
}

// ---------------------------------------------------------------------------
// serial_to_parallel: compact each band's timesteps to the front (per b).
// Positions are unique per (b,band) so plain stores onto the zeroed output.
// ---------------------------------------------------------------------------
__global__ void scatter_kernel(const unsigned char* __restrict__ s1,
                               const float* __restrict__ decx,
                               float* __restrict__ out) {
    const int b = blockIdx.x * blockDim.x + threadIdx.x;
    if (b >= B_) return;
    int cnt0 = 0, cnt1 = 0;
    for (int t = 0; t < T_; ++t) {
        const int   band = s1[((size_t)b * T_ + t) * 2 + 1] ? 1 : 0;
        const float v    = decx[(size_t)b * T_ + t];
        if (band) { out[(size_t)B_ * T_ + (size_t)b * T_ + cnt1] = v; ++cnt1; }
        else      { out[(size_t)b * T_ + cnt0] = v;               ++cnt0; }
    }
}

// ---------------------------------------------------------------------------
static inline size_t alignUp(size_t x, size_t a) { return (x + a - 1) & ~(a - 1); }

extern "C" void kernel_launch(void* const* d_in, const int* in_sizes, int n_in,
                              void* d_out, int out_size, void* d_ws, size_t ws_size,
                              hipStream_t stream) {
    const float*         dtime = (const float*)d_in[0];
    const unsigned char* s1    = (const unsigned char*)d_in[1];   // bool [B,T,2]
    const unsigned char* oh    = (const unsigned char*)d_in[2];   // bool [B,T]
    const float*         encz  = (const float*)d_in[3];
    const float*         W_ih  = (const float*)d_in[4];
    const float*         W_hh  = (const float*)d_in[5];
    const float*         b_ih  = (const float*)d_in[6];
    const float*         b_hh  = (const float*)d_in[7];
    const float*         W1    = (const float*)d_in[8];
    const float*         b1    = (const float*)d_in[9];
    const float*         W2    = (const float*)d_in[10];
    const float*         b2    = (const float*)d_in[11];
    float*               out   = (float*)d_out;

    // workspace carve-out (~10.8 MB)
    char* ws = (char*)d_ws;
    size_t off = 0;
    int* ctr = (int*)(ws + off);                 off = alignUp(off + T_ * sizeof(int), 256);
    float* hf32 = (float*)(ws + off);            off = alignUp(off + (size_t)B_ * H_ * 4, 256);
    __hip_bfloat16* hbf0 = (__hip_bfloat16*)(ws + off); off = alignUp(off + (size_t)B_ * H_ * 2, 256);
    __hip_bfloat16* hbf1 = (__hip_bfloat16*)(ws + off); off = alignUp(off + (size_t)B_ * H_ * 2, 256);
    __hip_bfloat16* WtHH = (__hip_bfloat16*)(ws + off); off = alignUp(off + (size_t)G3H * H_ * 2, 256);
    __hip_bfloat16* W1t  = (__hip_bfloat16*)(ws + off); off = alignUp(off + (size_t)H_ * H_ * 2, 256);
    float* partial = (float*)(ws + off);         off = alignUp(off + (size_t)T_ * NWG * B_ * 4, 256);
    float* decx    = (float*)(ws + off);         off = alignUp(off + (size_t)B_ * T_ * 4, 256);
    (void)ws_size; (void)n_in; (void)in_sizes; (void)out_size;

    prep_kernel<<<1024, 256, 0, stream>>>(encz, W_hh, W1, hf32, hbf0, WtHH, W1t, ctr);

    gru_kernel<<<NWG, 256, 0, stream>>>(dtime, s1, oh, W_ih, b_ih, b_hh, b1, W2,
                                        WtHH, W1t, hf32, hbf0, hbf1, partial, ctr);

    reduce_kernel<<<(B_ * T_ + 255) / 256, 256, 0, stream>>>(partial, b2, decx, out);

    scatter_kernel<<<2, 64, 0, stream>>>(s1, decx, out);
}
